// VectorAlarmworkRNN_82197084111096
// MI455X (gfx1250) — matrix operand order repaired and checked by reading, not compiled, not run on hardware
//
#include <hip/hip_runtime.h>
#include <hip/hip_bf16.h>

// ---------------------------------------------------------------------------
// RNN: U = X @ Win^T + b_in1        (parallel bf16-WMMA GEMM)
//      h = tanh(U_l + h @ Wrec^T)   (persistent kernel, grid-wide barrier,
//                                    Wrec column in LDS, 4-way WMMA ILP)
//      Y = tanh(h_L @ Wout^T + b_out)
// ---------------------------------------------------------------------------

typedef __attribute__((ext_vector_type(16))) __bf16 v16bf;
typedef __attribute__((ext_vector_type(8)))  __bf16 v8bf;
typedef __attribute__((ext_vector_type(8)))  float  v8f;
typedef __attribute__((ext_vector_type(8)))  float  float8;

static constexpr int Bsz  = 128;
static constexpr int Lsz  = 512;
static constexpr int DIN  = 512;
static constexpr int DH   = 1024;
static constexpr int DOUT = 256;
static constexpr int Mu   = Bsz * Lsz;     // 65536 rows of U
static constexpr int KSTEPS = DH / 32;     // 32 k-steps in the recurrence

// ---- WMMA fragment helpers (wave32, 16x16x32 bf16) -------------------------
// A (16x32, MxK), contiguous K to match B: lanes 0-15 -> row=lane,    K 0..15;
//                                          lanes 16-31 -> row=lane-16, K 16..31.
__device__ __forceinline__ v16bf a_frag_bf16(const __bf16* tile, int ld, int lane) {
  const int row = lane & 15;
  const int kb  = (lane >> 4) << 4;
  const __bf16* p = tile + (size_t)row * ld + kb;
  v8bf lo = *reinterpret_cast<const v8bf*>(p);
  v8bf hi = *reinterpret_cast<const v8bf*>(p + 8);
  v16bf a;
#pragma unroll
  for (int i = 0; i < 8; ++i) { a[i] = lo[i]; a[i + 8] = hi[i]; }
  return a;
}

__device__ __forceinline__ v16bf a_frag_f32(const float* tile, int ld, int lane) {
  const int row = lane & 15;
  const int kb  = (lane >> 4) << 4;
  const float* p = tile + (size_t)row * ld + kb;
  float8 lo = *reinterpret_cast<const float8*>(p);
  float8 hi = *reinterpret_cast<const float8*>(p + 8);
  v16bf a;
#pragma unroll
  for (int i = 0; i < 8; ++i) { a[i] = (__bf16)lo[i]; a[i + 8] = (__bf16)hi[i]; }
  return a;
}

// B (32x16, KxN) = W^T tile. W stored (N,K) row-major: lane's column is a
// contiguous run of 16 K-values (lanes 0-15: col=lane, K=0..15;
// lanes 16-31: col=lane-16, K=16..31). One 32-byte load per lane.
__device__ __forceinline__ v16bf b_frag_w(const __bf16* w_tile, int ld, int lane) {
  const int col   = lane & 15;
  const int khalf = (lane >> 4) << 4;
  return *reinterpret_cast<const v16bf*>(w_tile + (size_t)col * ld + khalf);
}

// ---- weight fp32 -> bf16 pre-conversion ------------------------------------
__global__ void cvt_f32_bf16_kernel(const float* __restrict__ src,
                                    __bf16* __restrict__ dst, int n) {
  int i = blockIdx.x * blockDim.x + threadIdx.x;
  if (i < n) dst[i] = (__bf16)src[i];
}

// ---- Kernel 1: U = X @ Win^T + b_in1  (M=65536, K=512, N=1024) -------------
__global__ void __launch_bounds__(128)
gemm_u_kernel(const float* __restrict__ X, const __bf16* __restrict__ Win,
              const float* __restrict__ bin, float* __restrict__ U) {
  const int lane  = threadIdx.x & 31;
  const int wave  = threadIdx.x >> 5;
  const int mtile = blockIdx.x;                         // 0..4095
  const int nbase = ((blockIdx.y * 4 + wave) * 4) * 16; // 0..960 step 64
  const float* Xrow = X + (size_t)mtile * 16 * DIN;

  v8f acc[4] = {};
  for (int k0 = 0; k0 < DIN; k0 += 32) {
    v16bf a = a_frag_f32(Xrow + k0, DIN, lane);
#pragma unroll
    for (int j = 0; j < 4; ++j) {
      v16bf b = b_frag_w(Win + (size_t)(nbase + j * 16) * DIN + k0, DIN, lane);
      acc[j] = __builtin_amdgcn_wmma_f32_16x16x32_bf16(
          false, a, false, b, (short)0, acc[j], false, false);
    }
  }
  const int mrow  = mtile * 16 + ((lane >> 4) << 3);  // M = mrow + r
  const int ncol0 = nbase + (lane & 15);
#pragma unroll
  for (int j = 0; j < 4; ++j) {
    const int n = ncol0 + j * 16;
    const float bias = bin[n];
#pragma unroll
    for (int r = 0; r < 8; ++r)
      U[(size_t)(mrow + r) * DH + n] = acc[j][r] + bias;
  }
}

// ---- Kernel 2: h2 = tanh(U[:, l=1, :]); also resets the grid barrier -------
__global__ void h_init_kernel(const float* __restrict__ U, __bf16* __restrict__ h,
                              unsigned* __restrict__ barrier_ctr) {
  if (blockIdx.x == 0 && threadIdx.x == 0) *barrier_ctr = 0u;
  int i = blockIdx.x * blockDim.x + threadIdx.x;   // 0..131071
  int b = i >> 10, j = i & (DH - 1);
  float u = U[((size_t)b * Lsz + 1) * DH + j];
  h[i] = (__bf16)tanhf(u);
}

// ---- Kernel 3: persistent recurrence --------------------------------------
// 64 blocks x 256 threads (8 waves). Block = one 16-wide N-tile column of
// Wrec (staged in LDS, fragment-major); wave = one of 8 M-tiles.
// 510 steps separated by a device-wide release/acquire barrier.
// 4 independent accumulators break the dependent WMMA chain (latency-bound).
__global__ void __launch_bounds__(256)
rec_persistent_kernel(const __bf16* __restrict__ Wrec,
                      const float* __restrict__ U,
                      __bf16* __restrict__ h0, __bf16* __restrict__ h1,
                      unsigned* __restrict__ ctr) {
  const int lane  = threadIdx.x & 31;
  const int mtile = threadIdx.x >> 5;   // 0..7
  const int ntile = blockIdx.x;         // 0..63
  const unsigned nblocks = gridDim.x;

  // Stage Wrec rows [ntile*16, +16) x K=1024 into LDS, fragment-major:
  // Bs[(ks*32 + lane)*16 + i] -> each lane's fragment is 32 contiguous bytes.
  __shared__ __bf16 Bs[KSTEPS * 32 * 16];   // 32 KB
  for (int slot = threadIdx.x; slot < KSTEPS * 32; slot += blockDim.x) {
    const int ks = slot >> 5, ln = slot & 31;
    const int col = ln & 15, khalf = (ln >> 4) << 4;
    const __bf16* src = Wrec + ((size_t)ntile * 16 + col) * DH + ks * 32 + khalf;
    *reinterpret_cast<v16bf*>(Bs + (size_t)slot * 16) =
        *reinterpret_cast<const v16bf*>(src);
  }
  __syncthreads();

  const int brow = mtile * 16 + ((lane >> 4) << 3);  // batch row = brow + r
  const int n    = ntile * 16 + (lane & 15);

  unsigned barriers_done = 0;
  for (int l = 2; l < Lsz; ++l) {
    const __bf16* hin  = (l & 1) ? h1 : h0;
    __bf16*       hout = (l & 1) ? h0 : h1;
    const __bf16* Arow = hin + (size_t)mtile * 16 * DH;

    // Prefetch this step's U slice (HBM-resident) so its latency overlaps
    // the 32-WMMA loop instead of stalling the epilogue.
#pragma unroll
    for (int r = 0; r < 8; ++r)
      __builtin_prefetch(&U[((size_t)(brow + r) * Lsz + l) * DH + n], 0, 1);

    v8f acc[4] = {};
    v16bf a = a_frag_bf16(Arow, DH, lane);           // prefetch k-step 0
#pragma unroll
    for (int ks = 0; ks < KSTEPS; ++ks) {
      v16bf a_next = (ks + 1 < KSTEPS) ? a_frag_bf16(Arow + (ks + 1) * 32, DH, lane) : a;
      v16bf b = *reinterpret_cast<const v16bf*>(Bs + (size_t)(ks * 32 + lane) * 16);
      acc[ks & 3] = __builtin_amdgcn_wmma_f32_16x16x32_bf16(
          false, a, false, b, (short)0, acc[ks & 3], false, false);
      a = a_next;
    }

#pragma unroll
    for (int r = 0; r < 8; ++r) {
      const int bidx = brow + r;
      float u = U[((size_t)bidx * Lsz + l) * DH + n];
      float s = (acc[0][r] + acc[1][r]) + (acc[2][r] + acc[3][r]);
      hout[(size_t)bidx * DH + n] = (__bf16)tanhf(s + u);
    }

    // ---- device-wide barrier (monotonic counter, reset by h_init) ----
    ++barriers_done;
    __syncthreads();
    if (threadIdx.x == 0) {
      __hip_atomic_fetch_add(ctr, 1u, __ATOMIC_RELEASE, __HIP_MEMORY_SCOPE_AGENT);
      const unsigned target = barriers_done * nblocks;
      while (__hip_atomic_load(ctr, __ATOMIC_ACQUIRE, __HIP_MEMORY_SCOPE_AGENT) < target)
        __builtin_amdgcn_s_sleep(1);
    }
    __syncthreads();
  }
}

// ---- Kernel 4: Y = tanh(h_L @ Wout^T + b_out)  (M=128, K=1024, N=256) ------
__global__ void __launch_bounds__(128)
out_kernel(const __bf16* __restrict__ h, const __bf16* __restrict__ Wout,
           const float* __restrict__ bout, float* __restrict__ Y) {
  const int lane  = threadIdx.x & 31;
  const int wave  = threadIdx.x >> 5;
  const int tile  = blockIdx.x * 4 + wave;  // 0..127
  const int mtile = tile >> 4;              // 0..7
  const int ntile = tile & 15;              // 0..15
  const __bf16* Arow = h    + (size_t)mtile * 16 * DH;
  const __bf16* Wn   = Wout + (size_t)ntile * 16 * DH;

  v8f acc[4] = {};
  v16bf a = a_frag_bf16(Arow, DH, lane);
#pragma unroll
  for (int ks = 0; ks < KSTEPS; ++ks) {
    v16bf a_next = (ks + 1 < KSTEPS) ? a_frag_bf16(Arow + (ks + 1) * 32, DH, lane) : a;
    v16bf b = b_frag_w(Wn + ks * 32, DH, lane);
    acc[ks & 3] = __builtin_amdgcn_wmma_f32_16x16x32_bf16(
        false, a, false, b, (short)0, acc[ks & 3], false, false);
    a = a_next;
  }
  const int brow = mtile * 16 + ((lane >> 4) << 3);
  const int n    = ntile * 16 + (lane & 15);
  const float bias = bout[n];
#pragma unroll
  for (int r = 0; r < 8; ++r) {
    float s = (acc[0][r] + acc[1][r]) + (acc[2][r] + acc[3][r]);
    Y[(size_t)(brow + r) * DOUT + n] = tanhf(s + bias);
  }
}

// ---------------------------------------------------------------------------
extern "C" void kernel_launch(void* const* d_in, const int* in_sizes, int n_in,
                              void* d_out, int out_size, void* d_ws, size_t ws_size,
                              hipStream_t stream) {
  const float* X      = (const float*)d_in[0];
  const float* W_in1  = (const float*)d_in[1];
  const float* b_in1  = (const float*)d_in[2];
  const float* W_rec1 = (const float*)d_in[3];
  // d_in[4..6] = W_in2 / b_in2 / W_rec2: dead code w.r.t. Y in the reference.
  const float* W_out  = (const float*)d_in[7];
  const float* b_out  = (const float*)d_in[8];

  // Workspace layout (all offsets 256B-aligned); total ~272.6 MB.
  char*     ws    = (char*)d_ws;
  size_t    off   = 0;
  float*    U     = (float*)(ws + off);    off += (size_t)Mu * DH * sizeof(float); // 256 MB
  __bf16*   WinB  = (__bf16*)(ws + off);   off += (size_t)DH * DIN * 2;            // 1 MB
  __bf16*   WrecB = (__bf16*)(ws + off);   off += (size_t)DH * DH * 2;             // 2 MB
  __bf16*   WoutB = (__bf16*)(ws + off);   off += (size_t)DOUT * DH * 2;           // 512 KB
  __bf16*   h0    = (__bf16*)(ws + off);   off += (size_t)Bsz * DH * 2;            // 256 KB
  __bf16*   h1    = (__bf16*)(ws + off);   off += (size_t)Bsz * DH * 2;            // 256 KB
  unsigned* ctr   = (unsigned*)(ws + off);

  // 1) weights -> bf16 once (L2-resident thereafter)
  cvt_f32_bf16_kernel<<<(DH * DIN + 255) / 256, 256, 0, stream>>>(W_in1, WinB, DH * DIN);
  cvt_f32_bf16_kernel<<<(DH * DH  + 255) / 256, 256, 0, stream>>>(W_rec1, WrecB, DH * DH);
  cvt_f32_bf16_kernel<<<(DOUT * DH + 255) / 256, 256, 0, stream>>>(W_out, WoutB, DOUT * DH);

  // 2) U = X @ Win^T + b_in1
  gemm_u_kernel<<<dim3(4096, 4), 128, 0, stream>>>(X, WinB, b_in1, U);

  // 3) h after l=1 is tanh(U_1) (l=0 result unused; l=1 ignores recurrence).
  //    Also resets the grid-barrier counter (deterministic per launch).
  h_init_kernel<<<(Bsz * DH) / 256, 256, 0, stream>>>(U, h0, ctr);

  // 4) all 510 recurrent steps in ONE persistent launch.
  //    Parity: l=2 reads h0 -> ... -> l=511 (odd) writes h0. Final state in h0.
  rec_persistent_kernel<<<64, 256, 0, stream>>>(WrecB, U, h0, h1, ctr);

  // 5) Y = tanh(h_L @ Wout^T + b_out)
  out_kernel<<<32, 128, 0, stream>>>(h0, WoutB, b_out, (float*)d_out);
}
